// HeteroHGATConv_90022514524491
// MI455X (gfx1250) — matrix operand order repaired and checked by reading, not compiled, not run on hardware
//
#include <hip/hip_runtime.h>
#include <hip/hip_bf16.h>
#include <stdint.h>

typedef __attribute__((ext_vector_type(16))) _Float16 v16h;
typedef __attribute__((ext_vector_type(8)))  float    v8f;
typedef int v4i_vec __attribute__((vector_size(16)));   // matches builtin param type

#define D_IN   256
#define D_OUT  256
#define N_EDGE 20000

#define AS1 __attribute__((address_space(1)))
#define AS3 __attribute__((address_space(3)))

#if __has_builtin(__builtin_amdgcn_global_load_async_to_lds_b128)
#define HAVE_ASYNC_LDS 1
#else
#define HAVE_ASYNC_LDS 0
#endif

// ---------------------------------------------------------------------------
// Weight repack: W (f32, row-major [K=256][N=256]) -> f16 WMMA B fragments.
// Fragment f = kt*16 + nt covers K in [kt*32, kt*32+32), N in [nt*16, nt*16+16).
// CDNA5 16-bit B 32x16 layout: lane l holds column n = l&15; half j (0..15)
// holds k = kt*32 + 8*(l>>4) + j + (j>=8?8:0).  Fragment stored lane-major, 16 halves/lane.
// ---------------------------------------------------------------------------
__global__ void repack_w(const float* __restrict__ W, _Float16* __restrict__ Bp) {
    int gid  = blockIdx.x * blockDim.x + threadIdx.x;   // 0..4095
    int f    = gid >> 5;
    int lane = gid & 31;
    int kt   = f >> 4;
    int nt   = f & 15;
    int n     = nt * 16 + (lane & 15);
    int kbase = kt * 32 + ((lane >> 4) << 3);
    _Float16* dst = Bp + ((size_t)(f * 32 + lane) << 4);
#pragma unroll
    for (int j = 0; j < 16; ++j)
        dst[j] = (_Float16)W[(size_t)(kbase + j + ((j & 8) ? 8 : 0)) * D_OUT + n];
}

// A fragment from LDS row-major f16 panel, ISA 16-bit A layout:
// lane l -> row rowbase + (l&15); dword v: k = kt*32 + 2v + (v<4?0:8) + (l>=16?8:0)
__device__ __forceinline__ void load_afrag(const _Float16* lh, int rowbase, int kt,
                                           int lane, uint32_t u[8]) {
    const uint32_t* lhu = (const uint32_t*)lh;
    const int row = rowbase + (lane & 15);
    const int lh8 = (lane >> 4) ? 8 : 0;
#pragma unroll
    for (int v = 0; v < 8; ++v) {
        int k = kt * 32 + 2 * v + ((v < 4) ? 0 : 8) + lh8;
        u[v] = lhu[(row * D_IN + k) >> 1];
    }
}

// Dual 16-row sub-panel GEMM: one B fragment load feeds two WMMAs when the
// two sub-panels use the same weight matrix (uniform scalar condition).
__device__ __forceinline__ void panel_gemm_dual(const _Float16* lh,
        const _Float16* __restrict__ BpA, const _Float16* __restrict__ BpB,
        int wave, int lane, v8f accA[4], v8f accB[4]) {
    const bool same = (BpA == BpB);
#pragma unroll
    for (int kt = 0; kt < 8; ++kt) {
        union { v16h h; uint32_t u[8]; } A0, A1;
        load_afrag(lh,  0, kt, lane, A0.u);
        load_afrag(lh, 16, kt, lane, A1.u);
#pragma unroll
        for (int nt = 0; nt < 4; ++nt) {
            size_t fo = ((size_t)((kt * 16 + wave * 4 + nt) * 32 + lane)) << 4;
            v16h B = *(const v16h*)(BpA + fo);
            accA[nt] = __builtin_amdgcn_wmma_f32_16x16x32_f16(
                false, A0.h, false, B, (short)0, accA[nt], false, false);
            if (!same) B = *(const v16h*)(BpB + fo);
            accB[nt] = __builtin_amdgcn_wmma_f32_16x16x32_f16(
                false, A1.h, false, B, (short)0, accB[nt], false, false);
        }
    }
}

// ---------------------------------------------------------------------------
// Fused: Xh = LN(lrelu(X@W{e|o}+b)) ; Xp = Xh@Wt + bt   (32 rows per block,
// 4 waves; wave w owns output columns [w*64, w*64+64) for all 32 rows)
// ---------------------------------------------------------------------------
__global__ __launch_bounds__(128) void fused_proj(
    const float* __restrict__ X, const int* __restrict__ teptr,
    const _Float16* __restrict__ BpE, const _Float16* __restrict__ BpO,
    const _Float16* __restrict__ BpT,
    const float* __restrict__ be, const float* __restrict__ ge, const float* __restrict__ bbe,
    const float* __restrict__ bo, const float* __restrict__ go, const float* __restrict__ bbo,
    const float* __restrict__ bt, float* __restrict__ Xp)
{
    __shared__ _Float16 lh[32 * D_IN];    // 16 KB: f16 A panel (both GEMMs)
    __shared__ float    lf[32 * D_OUT];   // 32 KB: staging + GEMM1 result
    __shared__ float    stats[64];        // per-row (mu, rsigma)

    const int tid  = threadIdx.x;
    const int wave = tid >> 5;
    const int lane = tid & 31;
    const int row0 = blockIdx.x * 32;

    const int  TE   = teptr[0];
    const bool isE0 = (row0      < TE);   // sub-panel 0 (TE is a multiple of 16)
    const bool isE1 = (row0 + 16 < TE);   // sub-panel 1
    const _Float16* BpA = isE0 ? BpE : BpO;
    const _Float16* BpB = isE1 ? BpE : BpO;
    const float* bA = isE0 ? be : bo;
    const float* bB = isE1 ? be : bo;

    // ---- Stage X panel (32x256 f32) -> f16 LDS ----
    const float4* X4 = (const float4*)(X + (size_t)row0 * D_IN);
#if HAVE_ASYNC_LDS
    {
        float4* lf4 = (float4*)lf;
#pragma unroll
        for (int it = 0; it < 16; ++it) {
            int fi = it * 128 + tid;
            __builtin_amdgcn_global_load_async_to_lds_b128(
                (AS1 v4i_vec*)(X4 + fi), (AS3 v4i_vec*)(lf4 + fi), 0, 0);
        }
#if __has_builtin(__builtin_amdgcn_s_wait_asynccnt)
        __builtin_amdgcn_s_wait_asynccnt(0);
#else
        asm volatile("s_wait_asynccnt 0" ::: "memory");
#endif
        __syncthreads();
#pragma unroll
        for (int it = 0; it < 16; ++it) {
            int fi = it * 128 + tid;
            float4 v = lf4[fi];
            int base = fi * 4;
            lh[base + 0] = (_Float16)v.x; lh[base + 1] = (_Float16)v.y;
            lh[base + 2] = (_Float16)v.z; lh[base + 3] = (_Float16)v.w;
        }
    }
#else
#pragma unroll
    for (int it = 0; it < 16; ++it) {
        int fi = it * 128 + tid;
        float4 v = X4[fi];
        int base = fi * 4;
        lh[base + 0] = (_Float16)v.x; lh[base + 1] = (_Float16)v.y;
        lh[base + 2] = (_Float16)v.z; lh[base + 3] = (_Float16)v.w;
    }
#endif
    __syncthreads();

    // ---- GEMM1: X @ W{e|o} ----
    v8f accA[4], accB[4];
#pragma unroll
    for (int i = 0; i < 4; ++i)
#pragma unroll
        for (int j = 0; j < 8; ++j) { accA[i][j] = 0.f; accB[i][j] = 0.f; }
    panel_gemm_dual(lh, BpA, BpB, wave, lane, accA, accB);

    // bias + leakyReLU(0.2) -> f32 LDS.  C layout: VGPR r -> M = r + 8*(lane>=16),
    // N = lane&15 within the tile.
    const int ncol = lane & 15;
    const int mrow = (lane >> 4) * 8;
#pragma unroll
    for (int nt = 0; nt < 4; ++nt) {
        int n = wave * 64 + nt * 16 + ncol;
        float biasA = bA[n], biasB = bB[n];
#pragma unroll
        for (int r = 0; r < 8; ++r) {
            float v = accA[nt][r] + biasA;
            v = v > 0.f ? v : 0.2f * v;
            lf[(mrow + r) * D_OUT + n] = v;
            v = accB[nt][r] + biasB;
            v = v > 0.f ? v : 0.2f * v;
            lf[(16 + mrow + r) * D_OUT + n] = v;
        }
    }
    __syncthreads();

    // ---- LayerNorm stats: 4 threads per row, shfl-reduce in groups of 4 ----
    const int rowln = tid >> 2, part = tid & 3;
    {
        const float* p = lf + rowln * D_OUT + part * 64;
        float s = 0.f, ss = 0.f;
#pragma unroll
        for (int i = 0; i < 64; ++i) { float x = p[i]; s += x; ss += x * x; }
#pragma unroll
        for (int o = 2; o > 0; o >>= 1) {
            s  += __shfl_xor(s,  o, 4);
            ss += __shfl_xor(ss, o, 4);
        }
        if (part == 0) {
            float mu  = s  * (1.f / D_OUT);
            float var = ss * (1.f / D_OUT) - mu * mu;
            stats[rowln * 2 + 0] = mu;
            stats[rowln * 2 + 1] = rsqrtf(var + 1e-5f);
        }
    }
    __syncthreads();

    // ---- Normalize * g + b -> f16 LDS panel (input of GEMM2) ----
    {
        float mu = stats[rowln * 2 + 0], rs = stats[rowln * 2 + 1];
        bool e = (row0 + rowln) < TE;
        const float* gg = e ? ge : go;
        const float* bb = e ? bbe : bbo;
        int c0 = part * 64;
#pragma unroll
        for (int i = 0; i < 64; ++i) {
            int n = c0 + i;
            float x = (lf[rowln * D_OUT + n] - mu) * rs * gg[n] + bb[n];
            lh[rowln * D_IN + n] = (_Float16)x;
        }
    }
    __syncthreads();

    // ---- GEMM2: Xh @ Wt + bt -> Xp (same Wt for both sub-panels: full B reuse)
    v8f acc2A[4], acc2B[4];
#pragma unroll
    for (int i = 0; i < 4; ++i)
#pragma unroll
        for (int j = 0; j < 8; ++j) { acc2A[i][j] = 0.f; acc2B[i][j] = 0.f; }
    panel_gemm_dual(lh, BpT, BpT, wave, lane, acc2A, acc2B);

#pragma unroll
    for (int nt = 0; nt < 4; ++nt) {
        int n = wave * 64 + nt * 16 + ncol;
        float bias = bt[n];
#pragma unroll
        for (int r = 0; r < 8; ++r) {
            Xp[(size_t)(row0 +      mrow + r) * D_OUT + n] = acc2A[nt][r] + bias;
            Xp[(size_t)(row0 + 16 + mrow + r) * D_OUT + n] = acc2B[nt][r] + bias;
        }
    }
}

// ------------------------- segment / attention kernels ---------------------

__global__ void k_zero(uint4* p, long n16) {
    long i = (long)blockIdx.x * blockDim.x + threadIdx.x;
    if (i < n16) p[i] = make_uint4(0u, 0u, 0u, 0u);
}

// v2e sum + degree: 4 pairs per 256-thread block, float4 per thread.
__global__ void k_ysum(const int* __restrict__ pv, const int* __restrict__ pe,
                       const float* __restrict__ Xp, float* __restrict__ Y,
                       float* __restrict__ deg, int nnz) {
    int p = blockIdx.x * 4 + (threadIdx.x >> 6);
    if (p >= nnz) return;
    int t = threadIdx.x & 63;
    int v = pv[p], e = pe[p];
    float4 xv = *(const float4*)(Xp + (size_t)v * D_OUT + t * 4);
    float* yp = Y + (size_t)e * D_OUT + t * 4;
    atomicAdd(yp + 0, xv.x); atomicAdd(yp + 1, xv.y);
    atomicAdd(yp + 2, xv.z); atomicAdd(yp + 3, xv.w);
    if (t == 0) atomicAdd(&deg[e], 1.0f);
}

__global__ void k_yalpha(float* __restrict__ Y, const float* __restrict__ deg,
                         const float* __restrict__ ae, float* __restrict__ alpha) {
    __shared__ float red[256];
    int e = blockIdx.x, d = threadIdx.x;
    float dg = deg[e]; dg = dg > 1.f ? dg : 1.f;
    float val = Y[(size_t)e * D_OUT + d] / dg;
    Y[(size_t)e * D_OUT + d] = val;
    red[d] = val * ae[d];
    __syncthreads();
    for (int o = 128; o > 0; o >>= 1) {
        if (d < o) red[d] += red[d + o];
        __syncthreads();
    }
    if (d == 0) alpha[e] = red[0];
}

__device__ __forceinline__ float pair_score(float a) {
    return (a > 0.f ? a : 0.2f * a) * 5.0f;
}
// order-preserving float<->uint for atomicMax over signed floats
__device__ __forceinline__ unsigned enc_f(float f) {
    unsigned b = __float_as_uint(f);
    return (b & 0x80000000u) ? ~b : (b | 0x80000000u);
}
__device__ __forceinline__ float dec_f(unsigned k) {
    if (k == 0u) return 0.f;      // empty segment -> m = 0 (matches reference)
    unsigned b = (k & 0x80000000u) ? (k & 0x7FFFFFFFu) : ~k;
    return __uint_as_float(b);
}

// score + segment max (also caches s into sbuf to avoid re-gathering alpha)
__global__ void k_smax(const int* __restrict__ pv, const int* __restrict__ pe,
                       const float* __restrict__ alpha, unsigned* __restrict__ mkey,
                       float* __restrict__ sbuf, int nnz) {
    int p = blockIdx.x * blockDim.x + threadIdx.x;
    if (p >= nnz) return;
    float s = pair_score(alpha[pe[p]]);
    sbuf[p] = s;
    atomicMax(&mkey[pv[p]], enc_f(s));
}

__global__ void k_den(const int* __restrict__ pv, const float* __restrict__ sbuf,
                      const unsigned* __restrict__ mkey, float* __restrict__ den, int nnz) {
    int p = blockIdx.x * blockDim.x + threadIdx.x;
    if (p >= nnz) return;
    int v = pv[p];
    atomicAdd(&den[v], expf(sbuf[p] - dec_f(mkey[v])));
}

// overwrite sbuf with final softmax weight w[p]
__global__ void k_w(const int* __restrict__ pv, const unsigned* __restrict__ mkey,
                    const float* __restrict__ den, float* __restrict__ sbuf, int nnz) {
    int p = blockIdx.x * blockDim.x + threadIdx.x;
    if (p >= nnz) return;
    int v = pv[p];
    float ex = expf(sbuf[p] - dec_f(mkey[v]));
    float dn = den[v]; dn = dn > 0.f ? dn : 1.f;
    sbuf[p] = ex / dn;
}

// e2v weighted scatter: 4 pairs per block, float4 per thread
__global__ void k_out(const int* __restrict__ pv, const int* __restrict__ pe,
                      const float* __restrict__ w, const float* __restrict__ Y,
                      float* __restrict__ out, int nnz) {
    int p = blockIdx.x * 4 + (threadIdx.x >> 6);
    if (p >= nnz) return;
    int t = threadIdx.x & 63;
    float wp = w[p];
    float4 yv = *(const float4*)(Y + (size_t)pe[p] * D_OUT + t * 4);
    float* op = out + (size_t)pv[p] * D_OUT + t * 4;
    atomicAdd(op + 0, wp * yv.x); atomicAdd(op + 1, wp * yv.y);
    atomicAdd(op + 2, wp * yv.z); atomicAdd(op + 3, wp * yv.w);
}

__global__ void k_elu4(float4* __restrict__ out, int n4) {
    int i = blockIdx.x * blockDim.x + threadIdx.x;
    if (i >= n4) return;
    float4 x = out[i];
    x.x = x.x > 0.f ? x.x : (expf(x.x) - 1.f);
    x.y = x.y > 0.f ? x.y : (expf(x.y) - 1.f);
    x.z = x.z > 0.f ? x.z : (expf(x.z) - 1.f);
    x.w = x.w > 0.f ? x.w : (expf(x.w) - 1.f);
    out[i] = x;
}

// ---------------------------------------------------------------------------
extern "C" void kernel_launch(void* const* d_in, const int* in_sizes, int n_in,
                              void* d_out, int out_size, void* d_ws, size_t ws_size,
                              hipStream_t stream) {
    const float* X   = (const float*)d_in[0];
    const int*   pv  = (const int*)  d_in[1];
    const int*   pe  = (const int*)  d_in[2];
    const int*   te  = (const int*)  d_in[3];
    const float* We  = (const float*)d_in[4];
    const float* be  = (const float*)d_in[5];
    const float* ge  = (const float*)d_in[6];
    const float* bbe = (const float*)d_in[7];
    const float* Wo  = (const float*)d_in[8];
    const float* bo  = (const float*)d_in[9];
    const float* go  = (const float*)d_in[10];
    const float* bbo = (const float*)d_in[11];
    const float* Wt  = (const float*)d_in[12];
    const float* bt  = (const float*)d_in[13];
    const float* ae  = (const float*)d_in[14];

    const int N   = in_sizes[0] / D_IN;   // 100000
    const int NNZ = in_sizes[1];          // 800000
    const int E   = N_EDGE;               // 20000

    // workspace layout (all offsets 32B-aligned)
    char* ws = (char*)d_ws;
    size_t off = 0;
    float*     Xp    = (float*)(ws + off);             off += (size_t)N * D_OUT * 4;
    size_t     offY  = off;
    float*     Y     = (float*)(ws + off);             off += (size_t)E * D_OUT * 4;
    float*     deg   = (float*)(ws + off);             off += (size_t)E * 4;
    float*     alpha = (float*)(ws + off);             off += (size_t)E * 4;
    unsigned*  mkey  = (unsigned*)(ws + off);          off += (size_t)N * 4;
    float*     den   = (float*)(ws + off);             off += (size_t)N * 4;
    size_t     offZEnd = off;
    float*     sbuf  = (float*)(ws + off);             off += (size_t)NNZ * 4;  // s then w
    _Float16*  BpE   = (_Float16*)(ws + off);          off += (size_t)D_IN * D_OUT * 2;
    _Float16*  BpO   = (_Float16*)(ws + off);          off += (size_t)D_IN * D_OUT * 2;
    _Float16*  BpT   = (_Float16*)(ws + off);          off += (size_t)D_IN * D_OUT * 2;

    float* out = (float*)d_out;

    // zero Y/deg/alpha/mkey/den region and d_out (accumulated via atomics)
    long z1 = (long)((offZEnd - offY) / 16);
    k_zero<<<(int)((z1 + 255) / 256), 256, 0, stream>>>((uint4*)(ws + offY), z1);
    long z2 = ((long)N * D_OUT * 4) / 16;
    k_zero<<<(int)((z2 + 255) / 256), 256, 0, stream>>>((uint4*)out, z2);

    // pack weights into WMMA B fragments (f16)
    repack_w<<<16, 256, 0, stream>>>(We, BpE);
    repack_w<<<16, 256, 0, stream>>>(Wo, BpO);
    repack_w<<<16, 256, 0, stream>>>(Wt, BpT);

    // fused projections + layernorm + theta GEMM (32 rows / block)
    fused_proj<<<N / 32, 128, 0, stream>>>(X, te, BpE, BpO, BpT,
                                           be, ge, bbe, bo, go, bbo, bt, Xp);

    const int nb  = (NNZ + 255) / 256;
    const int nb4 = (NNZ + 3) / 4;
    k_ysum  <<<nb4, 256, 0, stream>>>(pv, pe, Xp, Y, deg, NNZ);
    k_yalpha<<<E,   256, 0, stream>>>(Y, deg, ae, alpha);
    k_smax  <<<nb,  256, 0, stream>>>(pv, pe, alpha, mkey, sbuf, NNZ);
    k_den   <<<nb,  256, 0, stream>>>(pv, sbuf, mkey, den, NNZ);
    k_w     <<<nb,  256, 0, stream>>>(pv, mkey, den, sbuf, NNZ);
    k_out   <<<nb4, 256, 0, stream>>>(pv, pe, sbuf, Y, out, NNZ);
    k_elu4  <<<(N * D_OUT / 4 + 255) / 256, 256, 0, stream>>>((float4*)out, N * D_OUT / 4);
}
